// VectorAggModel_12386685682401
// MI455X (gfx1250) — matrix operand order repaired and checked by reading, not compiled, not run on hardware
//
#include <hip/hip_runtime.h>
#include <hip/hip_bf16.h>
#include <math.h>

// ---------------- problem constants ----------------
#define KB 7          // x branches
#define NB 8192       // batch
#define ED 500        // embed dim
#define HD 64         // heads
#define NJ 8          // branches incl. y-only
#define NCHUNK 16     // ceil(500/32) K-chunks of 32
#define EPS 1e-5f
#define INV_SQRT_E 0.044721359549995794f  // 1/sqrt(500)

typedef __bf16 bf16_t;
typedef bf16_t v16bf __attribute__((ext_vector_type(16)));
typedef float  v8f   __attribute__((ext_vector_type(8)));

__device__ __forceinline__ v8f wmma_bf16(v16bf a, v16bf b, v8f c) {
  // D = A(16x32 bf16) * B(32x16 bf16) + C(16x16 f32)
  return __builtin_amdgcn_wmma_f32_16x16x32_bf16(false, a, false, b, (short)0, c, false, false);
}

// ---------------- prep0: row layer-norm of params ----------------
// rows 0..447   : p_xy rows, length 1000 -> lnp_xy
// rows 448..511 : p_y  rows, length 500  -> lnp_y
__global__ void prep0_ln_params(const float* __restrict__ p_xy,
                                const float* __restrict__ p_y,
                                float* __restrict__ lnp_xy,
                                float* __restrict__ lnp_y) {
  int row = blockIdx.x;
  const float* in;
  float* out;
  int len;
  if (row < 448) { in = p_xy + (size_t)row * 1000; out = lnp_xy + (size_t)row * 1000; len = 1000; }
  else           { int r = row - 448; in = p_y + (size_t)r * 500; out = lnp_y + (size_t)r * 500; len = 500; }

  int t = threadIdx.x, w = t >> 5, lane = t & 31;
  float s = 0.f, q = 0.f;
  for (int e = t; e < len; e += 256) { float v = in[e]; s += v; q += v * v; }
  #pragma unroll
  for (int off = 16; off > 0; off >>= 1) { s += __shfl_xor(s, off, 32); q += __shfl_xor(q, off, 32); }
  __shared__ float red[16];
  if (lane == 0) { red[w] = s; red[8 + w] = q; }
  __syncthreads();
  float S = 0.f, Q = 0.f;
  #pragma unroll
  for (int i = 0; i < 8; i++) { S += red[i]; Q += red[8 + i]; }
  float inv = 1.0f / (float)len;
  float m = S * inv;
  float v = Q * inv - m * m;
  float r = rsqrtf(v + EPS);
  for (int e = t; e < len; e += 256) out[e] = (in[e] - m) * r;
}

// ---------------- statsk: sum/sumsq of each raw row ----------------
// 8 * 8192 rows of 500 (j<7 -> x_stack[j], j==7 -> y). One wave per row.
__global__ void statsk(const float* __restrict__ xs, const float* __restrict__ ys,
                       float* __restrict__ stats) {
  int w = threadIdx.x >> 5, lane = threadIdx.x & 31;
  int rIdx = blockIdx.x * 8 + w;               // 0 .. 65535
  int j = rIdx >> 13, b = rIdx & 8191;
  const float* row = (j < 7) ? (xs + ((size_t)j * NB + b) * ED) : (ys + (size_t)b * ED);
  float s = 0.f, q = 0.f;
  for (int e = lane; e < ED; e += 32) { float v = row[e]; s += v; q += v * v; }
  #pragma unroll
  for (int off = 16; off > 0; off >>= 1) { s += __shfl_xor(s, off, 32); q += __shfl_xor(q, off, 32); }
  if (lane == 0) { stats[(size_t)rIdx * 2] = s; stats[(size_t)rIdx * 2 + 1] = q; }
}

// ---------------- prep1: pack LN'd params into bf16 B-fragments ----------------
// B-matrix 32x16 per-lane packing used here (contiguous K, not ISA table order): lane = 16*Khalf + n holds 16
// contiguous K values (Khalf*16 + 0..15) of column n, packed 2/VGPR.
// WY: [chunk(16)][ntile(32)][lane(32)][16]  columns c=nt*16+n -> branch c/64, h=c%64
//     values = y-part of lnp_xy (e+500) for branch<7, lnp_y for branch 7
// WX: [k(7)][chunk(16)][ntile(4)][lane(32)][16]  h = nt*16+n, x-part of lnp_xy
__global__ void prep1_pack(const float* __restrict__ lnp_xy,
                           const float* __restrict__ lnp_y,
                           bf16_t* __restrict__ wy, bf16_t* __restrict__ wx) {
  int tid = blockIdx.x * blockDim.x + threadIdx.x;
  const int WYTOT = NCHUNK * 32 * 32 * 16;     // 262144
  const int WXTOT = KB * NCHUNK * 4 * 32 * 16; // 229376
  if (tid < WYTOT) {
    int j = tid & 15;
    int lane = (tid >> 4) & 31;
    int nt = (tid >> 9) & 31;
    int chunk = tid >> 14;
    int e = chunk * 32 + ((lane >> 4) << 4) + j;   // K index 0..511
    int c = nt * 16 + (lane & 15);
    int kb = c >> 6, h = c & 63;
    float val = 0.f;
    if (e < ED) val = (kb < 7) ? lnp_xy[((size_t)kb * HD + h) * 1000 + 500 + e]
                               : lnp_y[(size_t)h * ED + e];
    wy[tid] = (bf16_t)val;
  } else if (tid < WYTOT + WXTOT) {
    int t2 = tid - WYTOT;
    int j = t2 & 15;
    int lane = (t2 >> 4) & 31;
    int nt = (t2 >> 9) & 3;
    int chunk = (t2 >> 11) & 15;
    int k = t2 >> 15;
    int e = chunk * 32 + ((lane >> 4) << 4) + j;
    int h = nt * 16 + (lane & 15);
    float val = (e < ED) ? lnp_xy[((size_t)k * HD + h) * 1000 + e] : 0.f;
    wx[t2] = (bf16_t)val;
  }
}

// A-matrix 16x32 bf16 per-lane packing used here (contiguous K): lane (m = lane&15,
// half = lane>>4): elems 0..15 -> K = half*16 + 0..15 (matches B packing)
__device__ __forceinline__ v16bf load_a_frag(const float* __restrict__ row, int chunk, int half) {
  int b0 = chunk * 32 + half * 16;
  int b1 = b0 + 8;
  v16bf a;
  #pragma unroll
  for (int t = 0; t < 8; t++) {
    int e0 = b0 + t, e1 = b1 + t;
    a[t]     = (e0 < ED) ? (bf16_t)row[e0] : (bf16_t)0.0f;
    a[8 + t] = (e1 < ED) ? (bf16_t)row[e1] : (bf16_t)0.0f;
  }
  return a;
}

// ---------------- kernel 1: WMMA scores + softmax -> combine coeffs ----------------
// grid 512 blocks x 256 threads (8 waves). Block = 16 batch rows; wave k = branch k.
__global__ void score_k(const float* __restrict__ xs, const float* __restrict__ ys,
                        const bf16_t* __restrict__ wy, const bf16_t* __restrict__ wx,
                        const float* __restrict__ stats,
                        float* __restrict__ Cbuf, float* __restrict__ Dbuf) {
  __shared__ float sScores[NJ][16][HD];   // 32 KB
  __shared__ float sm500[NJ][16], sr500[NJ][16], sr1000[NJ][16];

  int t = threadIdx.x;
  int b0 = blockIdx.x * 16;

  // per-(branch,row) LN scales from precomputed sums
  if (t < 128) {
    int j = t >> 4, bl = t & 15, b = b0 + bl;
    float sx = stats[((size_t)j * NB + b) * 2];
    float qx = stats[((size_t)j * NB + b) * 2 + 1];
    float sy = stats[((size_t)7 * NB + b) * 2];
    float qy = stats[((size_t)7 * NB + b) * 2 + 1];
    float m5 = sx * (1.f / 500.f);
    float v5 = qx * (1.f / 500.f) - m5 * m5;
    float r5 = rsqrtf(v5 + EPS);
    sm500[j][bl] = m5;
    sr500[j][bl] = r5;
    float r10;
    if (j < 7) {
      float m10 = (sx + sy) * (1.f / 1000.f);
      float v10 = (qx + qy) * (1.f / 1000.f) - m10 * m10;
      r10 = rsqrtf(v10 + EPS);
    } else {
      r10 = r5;                       // y-only branch uses 500-dim LN
    }
    sr1000[j][bl] = r10;
  }

  int k = t >> 5;            // wave -> branch
  int lane = t & 31;
  int m = lane & 15, half = lane >> 4;

  const float* yrow = ys + (size_t)(b0 + m) * ED;
  const float* xrow = (k < 7) ? (xs + ((size_t)k * NB + (b0 + m)) * ED) : yrow;

  const v16bf* wyv = reinterpret_cast<const v16bf*>(wy);
  const v16bf* wxv = reinterpret_cast<const v16bf*>(wx);

  v8f acc[4];
  #pragma unroll
  for (int nt = 0; nt < 4; nt++) acc[nt] = (v8f)(0.f);

  for (int chunk = 0; chunk < NCHUNK; chunk++) {
    v16bf ay = load_a_frag(yrow, chunk, half);
    v16bf ax;
    if (k < 7) ax = load_a_frag(xrow, chunk, half);
    #pragma unroll
    for (int nt = 0; nt < 4; nt++) {
      int ntg = k * 4 + nt;                       // global ntile (h group) for this branch
      v16bf by = wyv[(size_t)(chunk * 32 + ntg) * 32 + lane];
      acc[nt] = wmma_bf16(ay, by, acc[nt]);
      if (k < 7) {
        v16bf bx = wxv[(size_t)(((k * NCHUNK + chunk) * 4 + nt)) * 32 + lane];
        acc[nt] = wmma_bf16(ax, bx, acc[nt]);
      }
    }
  }
  __syncthreads();   // sm/sr arrays ready (also separates score writes)

  // C/D layout: VGPR r, lane -> row m = r + 8*half, col n = lane&15
  #pragma unroll
  for (int nt = 0; nt < 4; nt++) {
    #pragma unroll
    for (int r = 0; r < 8; r++) {
      int mr = r + 8 * half;
      sScores[k][mr][nt * 16 + m] = acc[nt][r] * sr1000[k][mr] * INV_SQRT_E;
    }
  }
  __syncthreads();

  // softmax over the 8 branches per (b,h); emit combine coeffs
  for (int p = t; p < 16 * HD; p += 256) {
    int bl = p >> 6, h = p & 63;
    float s[NJ], mx = -3.0e38f;
    #pragma unroll
    for (int j = 0; j < NJ; j++) { s[j] = sScores[j][bl][h]; mx = fmaxf(mx, s[j]); }
    float sum = 0.f;
    #pragma unroll
    for (int j = 0; j < NJ; j++) { s[j] = expf(s[j] - mx); sum += s[j]; }
    float inv = 1.f / sum;
    float d = 0.f;
    size_t base = ((size_t)(b0 + bl) * HD + h);
    #pragma unroll
    for (int j = 0; j < NJ; j++) {
      float c = s[j] * inv * sr500[j][bl];
      Cbuf[base * NJ + j] = c;
      d += c * sm500[j][bl];
    }
    Dbuf[base] = d;
  }
}

// ---------------- kernel 2: combine + GeLU + row-LN + store ----------------
// 1 block per batch row b; 256 threads (8 waves); wave w handles h = w*8 .. w*8+7.
__global__ void out_k(const float* __restrict__ xs, const float* __restrict__ ys,
                      const float* __restrict__ Cbuf, const float* __restrict__ Dbuf,
                      float* __restrict__ out) {
  __shared__ float fx[NJ * ED];   // 16 KB raw rows
  __shared__ float sC[HD * NJ];
  __shared__ float sD[HD];

  int b = blockIdx.x;
  int t = threadIdx.x;

  for (int idx = t; idx < NJ * ED; idx += 256) {
    int j = idx / ED, e = idx - j * ED;
    fx[idx] = (j < 7) ? xs[((size_t)j * NB + b) * ED + e] : ys[(size_t)b * ED + e];
  }
  for (int idx = t; idx < HD * NJ; idx += 256) sC[idx] = Cbuf[(size_t)b * HD * NJ + idx];
  if (t < HD) sD[t] = Dbuf[(size_t)b * HD + t];
  __syncthreads();

  int w = t >> 5, lane = t & 31;
  for (int hi = 0; hi < 8; hi++) {
    int h = w * 8 + hi;
    float c[NJ];
    #pragma unroll
    for (int j = 0; j < NJ; j++) c[j] = sC[h * NJ + j];
    float d = sD[h];

    float vals[16];
    float s = 0.f, q = 0.f;
    #pragma unroll
    for (int i = 0; i < 16; i++) {
      int e = lane + 32 * i;
      float g = 0.f;
      if (e < ED) {
        float x = -d;
        #pragma unroll
        for (int j = 0; j < NJ; j++) x = fmaf(c[j], fx[j * ED + e], x);
        g = 0.5f * x * (1.f + erff(x * 0.70710678118654752f));   // exact GeLU
        s += g;
        q += g * g;
      }
      vals[i] = g;
    }
    #pragma unroll
    for (int off = 16; off > 0; off >>= 1) { s += __shfl_xor(s, off, 32); q += __shfl_xor(q, off, 32); }
    float m = s * (1.f / 500.f);
    float v = q * (1.f / 500.f) - m * m;
    float r = rsqrtf(v + EPS);
    size_t base = ((size_t)b * HD + h) * ED;
    #pragma unroll
    for (int i = 0; i < 16; i++) {
      int e = lane + 32 * i;
      if (e < ED) out[base + e] = (vals[i] - m) * r;
    }
  }
}

// ---------------- launch ----------------
extern "C" void kernel_launch(void* const* d_in, const int* in_sizes, int n_in,
                              void* d_out, int out_size, void* d_ws, size_t ws_size,
                              hipStream_t stream) {
  const float* x_stack = (const float*)d_in[0];   // [7,8192,1,500]
  const float* y       = (const float*)d_in[1];   // [8192,1,500]
  const float* p_xy    = (const float*)d_in[2];   // [7,64,1000]
  const float* p_y     = (const float*)d_in[3];   // [64,500]
  float* out = (float*)d_out;                     // [8192,64,500]

  // workspace carve-up (all offsets keep 32B+ alignment)
  char* w = (char*)d_ws;
  float*  lnp_xy = (float*)w;                          // 448000 f32
  float*  lnp_y  = lnp_xy + 448000;                    // 32000  f32
  bf16_t* wy     = (bf16_t*)(lnp_y + 32000);           // 262144 bf16
  bf16_t* wx     = wy + 262144;                        // 229376 bf16
  float*  stats  = (float*)(wx + 229376);              // 8*8192*2 f32
  float*  Cbuf   = stats + (size_t)NJ * NB * 2;        // 8192*64*8 f32
  float*  Dbuf   = Cbuf + (size_t)NB * HD * NJ;        // 8192*64 f32

  prep0_ln_params<<<dim3(512), dim3(256), 0, stream>>>(p_xy, p_y, lnp_xy, lnp_y);
  statsk<<<dim3(8192), dim3(256), 0, stream>>>(x_stack, y, stats);
  prep1_pack<<<dim3(1920), dim3(256), 0, stream>>>(lnp_xy, lnp_y, wy, wx);
  score_k<<<dim3(NB / 16), dim3(256), 0, stream>>>(x_stack, y, wy, wx, stats, Cbuf, Dbuf);
  out_k<<<dim3(NB), dim3(256), 0, stream>>>(x_stack, y, Cbuf, Dbuf, out);
}
